// DynamicConv2d_28321014350080
// MI455X (gfx1250) — matrix operand order repaired and checked by reading, not compiled, not run on hardware
//
#include <hip/hip_runtime.h>
#include <hip/hip_bf16.h>

typedef __attribute__((ext_vector_type(16))) _Float16 v16h;
typedef __attribute__((ext_vector_type(8)))  float    v8f;

// ---------------------------------------------------------------------------
// Sizes (fixed by the reference): B=64, C=256, H=W=48, G=4, K=3, Cr=64
// T matrix: rows 0..575 = pooled (b*9+kk), rows 576..639 = gap (b)
// ---------------------------------------------------------------------------

// Kernel 0: convert weights to f16, fold BN into scale/shift
__global__ __launch_bounds__(256) void dc_prep(
    const float* __restrict__ w1, const float* __restrict__ w2,
    const float* __restrict__ gam, const float* __restrict__ bet,
    const float* __restrict__ mean, const float* __restrict__ var,
    _Float16* __restrict__ w1h, _Float16* __restrict__ w2h,
    float* __restrict__ bn_inv, float* __restrict__ bn_shift)
{
    const int idx = blockIdx.x * 256 + threadIdx.x;
    if (idx < 16384) w1h[idx] = (_Float16)w1[idx];
    if (idx < 65536) w2h[idx] = (_Float16)w2[idx];
    if (idx < 64) {
        float inv = gam[idx] * rsqrtf(var[idx] + 1e-5f);
        bn_inv[idx]   = inv;
        bn_shift[idx] = bet[idx] - mean[idx] * inv;
    }
}

// Kernel 1: per-(b,c) plane -> 9 block means (16x16) + global mean, into T
__global__ __launch_bounds__(160) void dc_pool(
    const float* __restrict__ x, float* __restrict__ proj_in)
{
    __shared__ float segs[144];
    __shared__ float bsum[9];
    const int plane = blockIdx.x;          // b*256 + c
    const int tid   = threadIdx.x;
    if (tid < 144) {
        const int r = tid / 3, s = tid % 3;
        const float4* p = (const float4*)(x + (size_t)plane * 2304 + r * 48 + s * 16);
        float4 a = p[0], b = p[1], c = p[2], d = p[3];
        segs[tid] = (a.x + a.y + a.z + a.w) + (b.x + b.y + b.z + b.w) +
                    (c.x + c.y + c.z + c.w) + (d.x + d.y + d.z + d.w);
    }
    __syncthreads();
    const int b = plane >> 8, c = plane & 255;
    if (tid < 9) {
        const int bh = tid / 3, bw = tid % 3;
        float s = 0.f;
        #pragma unroll
        for (int rr = 0; rr < 16; ++rr) s += segs[(bh * 16 + rr) * 3 + bw];
        bsum[tid] = s;
        proj_in[(b * 9 + tid) * 256 + c] = s * (1.0f / 256.0f);
    }
    __syncthreads();
    if (tid == 0) {
        float tot = 0.f;
        #pragma unroll
        for (int k = 0; k < 9; ++k) tot += bsum[k];
        proj_in[(576 + b) * 256 + c] = tot * (1.0f / 2304.0f);
    }
}

// Kernel 2: WMMA projection (T x W1^T -> BN -> GELU -> x W2^T + b2),
// register softmax over G, blend dyn_weight / dyn_bias.
// One block = 8 waves handles a 16-row tile of T.
__global__ __launch_bounds__(256) void dc_proj(
    const float*    __restrict__ proj_in,   // [640,256]
    const _Float16* __restrict__ w1h,       // [64,256]   (row o, col c)
    const _Float16* __restrict__ w2h,       // [1024,64]
    const float*    __restrict__ bn_inv,    // [64]
    const float*    __restrict__ bn_shift,  // [64]
    const float*    __restrict__ b2,        // [1024]
    const float*    __restrict__ dyn_w,     // [4,256,9]
    const float*    __restrict__ dyn_b,     // [4,256]
    float*          __restrict__ weight_out,// [64,256,9]
    float*          __restrict__ bias_out)  // [64*256]
{
    __shared__ _Float16 Tt[16][264];   // activation tile, f16
    __shared__ _Float16 Y1[16][72];    // GELU output, f16

    const int tid  = threadIdx.x;
    const int wave = tid >> 5;
    const int lane = tid & 31;
    const int half = lane >> 4;
    const int l16  = lane & 15;
    const int row0 = blockIdx.x * 16;

    // Load T tile to LDS as f16
    for (int idx = tid; idx < 16 * 256; idx += 256) {
        const int m = idx >> 8, c = idx & 255;
        Tt[m][c] = (_Float16)proj_in[(row0 + m) * 256 + c];
    }
    __syncthreads();

    // -------- Stage A: Y1[16,64] = T(16x256) x W1^T(256x64), BN + GELU -----
    if (wave < 4) {
        const int N = wave * 16 + l16;         // output channel 0..63
        v8f acc = {};
        #pragma unroll
        for (int kt = 0; kt < 8; ++kt) {
            const int K0 = kt * 32;
            v16h a, bf;
            #pragma unroll
            for (int i = 0; i < 8; ++i) {
                const int kk = K0 + half * 16 + i * 2;
                a[2 * i]     = Tt[l16][kk];
                a[2 * i + 1] = Tt[l16][kk + 1];
            }
            const _Float16* bp = w1h + N * 256 + K0 + half * 16;
            #pragma unroll
            for (int j = 0; j < 16; ++j) bf[j] = bp[j];
            acc = __builtin_amdgcn_wmma_f32_16x16x32_f16(
                false, a, false, bf, (short)0, acc, false, false);
        }
        const float inv = bn_inv[N], shf = bn_shift[N];
        #pragma unroll
        for (int r = 0; r < 8; ++r) {
            const int M = r + half * 8;
            const float y = acc[r] * inv + shf;
            const float g = 0.5f * y * (1.0f + erff(y * 0.70710678118654752f));
            Y1[M][N] = (_Float16)g;
        }
    }
    __syncthreads();

    // -------- Stage B: Y2 = Y1(16x64) x W2^T(64x1024) + b2; softmax(G) ----
    // A fragments shared across all output tiles: build once.
    v16h aK[2];
    #pragma unroll
    for (int kt = 0; kt < 2; ++kt)
        #pragma unroll
        for (int i = 0; i < 8; ++i) {
            const int kk = kt * 32 + half * 16 + i * 2;
            aK[kt][2 * i]     = Y1[l16][kk];
            aK[kt][2 * i + 1] = Y1[l16][kk + 1];
        }

    const bool is_pooled = (row0 < 576);
    #pragma unroll
    for (int j = 0; j < 2; ++j) {
        const int ct = wave * 2 + j;           // channel tile 0..15
        const int c  = ct * 16 + l16;          // channel 0..255 (fixed per lane)
        v8f acc[4] = {};
        float b2c[4];
        #pragma unroll
        for (int g = 0; g < 4; ++g) {
            const int N = g * 256 + c;
            b2c[g] = b2[N];
            #pragma unroll
            for (int kt = 0; kt < 2; ++kt) {
                v16h bf;
                const _Float16* bp = w2h + N * 64 + kt * 32 + half * 16;
                #pragma unroll
                for (int q = 0; q < 16; ++q) bf[q] = bp[q];
                acc[g] = __builtin_amdgcn_wmma_f32_16x16x32_f16(
                    false, aK[kt], false, bf, (short)0, acc[g], false, false);
            }
        }
        // register softmax over g, then blend
        #pragma unroll
        for (int r = 0; r < 8; ++r) {
            const int M    = r + half * 8;
            const int grow = row0 + M;
            const float v0 = acc[0][r] + b2c[0];
            const float v1 = acc[1][r] + b2c[1];
            const float v2 = acc[2][r] + b2c[2];
            const float v3 = acc[3][r] + b2c[3];
            const float mx = fmaxf(fmaxf(v0, v1), fmaxf(v2, v3));
            const float e0 = __expf(v0 - mx), e1 = __expf(v1 - mx);
            const float e2 = __expf(v2 - mx), e3 = __expf(v3 - mx);
            const float rs = 1.0f / (e0 + e1 + e2 + e3);
            if (is_pooled) {
                const int bb = grow / 9, kk = grow % 9;
                const float wsum =
                    (e0 * dyn_w[(0 * 256 + c) * 9 + kk] +
                     e1 * dyn_w[(1 * 256 + c) * 9 + kk] +
                     e2 * dyn_w[(2 * 256 + c) * 9 + kk] +
                     e3 * dyn_w[(3 * 256 + c) * 9 + kk]) * rs;
                weight_out[(bb * 256 + c) * 9 + kk] = wsum;
            } else {
                const int bb = grow - 576;
                const float bsum =
                    (e0 * dyn_b[0 * 256 + c] + e1 * dyn_b[1 * 256 + c] +
                     e2 * dyn_b[2 * 256 + c] + e3 * dyn_b[3 * 256 + c]) * rs;
                bias_out[bb * 256 + c] = bsum;
            }
        }
    }
}

// Kernel 3: per-(b,c) depthwise 3x3, padding 1, + bias. One block per plane.
__global__ __launch_bounds__(256) void dc_dwconv(
    const float* __restrict__ x, const float* __restrict__ weight,
    const float* __restrict__ bias, float* __restrict__ out)
{
    __shared__ float tile[2304];
    const int    plane = blockIdx.x;
    const int    tid   = threadIdx.x;
    const size_t pbase = (size_t)plane * 2304;

    const float4* src  = (const float4*)(x + pbase);
    float4*       dst4 = (float4*)tile;
    dst4[tid]       = src[tid];
    dst4[tid + 256] = src[tid + 256];
    if (tid < 64) dst4[tid + 512] = src[tid + 512];

    const float* wp = weight + plane * 9;
    const float w00 = wp[0], w01 = wp[1], w02 = wp[2];
    const float w10 = wp[3], w11 = wp[4], w12 = wp[5];
    const float w20 = wp[6], w21 = wp[7], w22 = wp[8];
    const float bb  = bias[plane];
    __syncthreads();

    #pragma unroll
    for (int i = 0; i < 9; ++i) {
        const int p = tid + i * 256;           // 9*256 == 2304 exactly
        const int h = p / 48, w = p % 48;
        const float* r1 = tile + p;
        float acc = bb + w11 * r1[0];
        const bool wlo = (w > 0), whi = (w < 47);
        if (wlo) acc += w10 * r1[-1];
        if (whi) acc += w12 * r1[1];
        if (h > 0) {
            const float* r0 = r1 - 48;
            acc += w01 * r0[0];
            if (wlo) acc += w00 * r0[-1];
            if (whi) acc += w02 * r0[1];
        }
        if (h < 47) {
            const float* r2 = r1 + 48;
            acc += w21 * r2[0];
            if (wlo) acc += w20 * r2[-1];
            if (whi) acc += w22 * r2[1];
        }
        out[pbase + p] = acc;
    }
}

extern "C" void kernel_launch(void* const* d_in, const int* in_sizes, int n_in,
                              void* d_out, int out_size, void* d_ws, size_t ws_size,
                              hipStream_t stream) {
    const float* x    = (const float*)d_in[0];
    const float* w1   = (const float*)d_in[1];
    const float* gam  = (const float*)d_in[2];
    const float* bet  = (const float*)d_in[3];
    const float* mean = (const float*)d_in[4];
    const float* var  = (const float*)d_in[5];
    const float* w2   = (const float*)d_in[6];
    const float* b2   = (const float*)d_in[7];
    const float* dynw = (const float*)d_in[8];
    const float* dynb = (const float*)d_in[9];
    float* out = (float*)d_out;

    // workspace layout (256B-aligned offsets), ~1.41 MB total
    char* ws = (char*)d_ws;
    float*    proj_in    = (float*)   (ws + 0);        // 640*256 f32
    float*    weight_out = (float*)   (ws + 655360);   // 64*256*9 f32
    float*    bias_out   = (float*)   (ws + 1245184);  // 64*256 f32
    _Float16* w1h        = (_Float16*)(ws + 1310720);  // 16384 f16
    _Float16* w2h        = (_Float16*)(ws + 1343488);  // 65536 f16
    float*    bn_inv     = (float*)   (ws + 1474560);  // 64 f32
    float*    bn_shift   = (float*)   (ws + 1474816);  // 64 f32

    dc_prep<<<256, 256, 0, stream>>>(w1, w2, gam, bet, mean, var,
                                     w1h, w2h, bn_inv, bn_shift);
    dc_pool<<<16384, 160, 0, stream>>>(x, proj_in);
    dc_proj<<<40, 256, 0, stream>>>(proj_in, w1h, w2h, bn_inv, bn_shift, b2,
                                    dynw, dynb, weight_out, bias_out);
    dc_dwconv<<<16384, 256, 0, stream>>>(x, weight_out, bias_out, out);
}
